// ESNGenerator_42786464203602
// MI455X (gfx1250) — matrix operand order repaired and checked by reading, not compiled, not run on hardware
//
#include <hip/hip_runtime.h>
#include <hip/hip_fp16.h>
#include <hip/hip_bf16.h>

// ESN rollout on gfx1250 (MI455X): wave32, WMMA f32_16x16x32_f16.
//
//   X_t = tanh(X_{t-1} A^T + xi_t C^T)   (N=256, H=512, M=64)
//   Z_t = X_t W^T + eta_t                 (D=64), emit t >= washout
//
// Grid: 16 blocks (16 batch rows each). Block: 1024 threads = 32 waves.
// Wave w owns H-columns [16w,16w+16). State double-buffered in LDS.
// Z readout K-split across waves 0..7, combined via LDS partials.
// A laundered (always-zero) scalar offset is added to the weight bases each
// step: defeats LICM (round-1 asm showed 272 hoisted VGPRs spilling to
// scratch) while keeping the kernarg pointer chain visible to address-space
// inference so the weight stream lowers to GLOBAL (saddr) loads, not FLAT
// (round-2 asm showed flat_load_b128 entangling LOADcnt with DScnt).

typedef __attribute__((ext_vector_type(16))) _Float16 v16h;
typedef __attribute__((ext_vector_type(8)))  _Float16 v8h;
typedef __attribute__((ext_vector_type(8)))  float    v8f;
typedef __attribute__((ext_vector_type(4)))  float    v4f;

#define Hdim     512
#define Mdim     64
#define Ddim     64
#define Nbatch   256
#define T_OUT    2048
#define WASHOUT_ 64
#define TOTAL_T  (T_OUT + WASHOUT_)

// LDS row pad: 520*2B = 1040B row stride -> 4-bank rotation per row,
// rows 0..15 land on disjoint bank quads for the 16B fragment reads.
#define XROW 520

// ---------------------------------------------------------------------------
// One-time fp32 -> fp16 conversion of A (HxH), C (HxM), W (DxH) into d_ws.
// All three are consumed row-major as-is (B-operand = two 8-half K runs, 16 apart).
// ---------------------------------------------------------------------------
__global__ void esn_convert_f16(const float* __restrict__ A,
                                const float* __restrict__ C,
                                const float* __restrict__ W,
                                _Float16* __restrict__ Af16,
                                _Float16* __restrict__ Cf16,
                                _Float16* __restrict__ Wf16) {
  const int i = blockIdx.x * blockDim.x + threadIdx.x;
  if (i < Hdim * Hdim) Af16[i] = (_Float16)A[i];
  if (i < Hdim * Mdim) Cf16[i] = (_Float16)C[i];
  if (i < Ddim * Hdim) Wf16[i] = (_Float16)W[i];
}

// Build a v16h WMMA operand from two contiguous 8-half loads.
__device__ __forceinline__ v16h pack16(v8h lo, v8h hi) {
  v16h r;
#pragma unroll
  for (int e = 0; e < 8; ++e) { r[e] = lo[e]; r[8 + e] = hi[e]; }
  return r;
}

// ---------------------------------------------------------------------------
// Persistent per-batch-tile rollout kernel.
// ---------------------------------------------------------------------------
__global__ __launch_bounds__(1024)
void esn_rollout(const _Float16* __restrict__ Af16,
                 const _Float16* __restrict__ Cf16,
                 const _Float16* __restrict__ Wf16,
                 const float* __restrict__ xi,
                 const float* __restrict__ eta,
                 float* __restrict__ out) {
  __shared__ __align__(16) _Float16 Xbuf[2][16][XROW];
  __shared__ __align__(16) float    Zpart[4][16][16];  // K-upper partials

  const unsigned tid  = threadIdx.x;
  const unsigned lane = tid & 31u;   // wave32
  const unsigned wave = tid >> 5;    // 0..31
  const unsigned lo16 = lane & 15u;  // WMMA row (A-op) / col (B,D-op) select
  const unsigned hi   = lane >> 4;   // lane-half select per ISA layout
  const unsigned n0   = blockIdx.x * 16u;  // batch tile base
  const unsigned h0   = wave * 16u;        // H-column tile of this wave

  // X_{-1} = 0
  for (unsigned i = tid; i < 2u * 16u * XROW; i += blockDim.x)
    ((_Float16*)Xbuf)[i] = (_Float16)0.0f;
  __syncthreads();

  // Laundered always-zero element offset; redefined every iteration so the
  // weight loads are loop-variant (no LICM/spill), but the base pointers
  // remain provably global (saddr GLOBAL loads).
  size_t lz = 0;

  for (int t = 0; t < TOTAL_T; ++t) {
    asm volatile("" : "+s"(lz));
    const _Float16* __restrict__ Ab = Af16 + lz;
    const _Float16* __restrict__ Cb = Cf16 + lz;
    const _Float16* __restrict__ Wb = Wf16 + lz;
    const int p = t & 1;            // read buffer; write p^1

    // Warm next step's xi rows (global_prefetch_b8), off the critical path.
    if (t + 1 < TOTAL_T) {
      const float* xnext =
          xi + ((size_t)(n0 + lo16) * TOTAL_T + (size_t)(t + 1)) * Mdim +
          (size_t)hi * 32;
      __builtin_prefetch(xnext, 0, 1);
    }

    v8f acc = {};

    // ---- xi_t C^T contribution: K = M = 64, two K=32 chunks -------------
#pragma unroll
    for (int mc = 0; mc < 2; ++mc) {
      const int kb = mc * 32 + (int)hi * 8;     // A-op elems 0..7 K-base
      const float* xr =
          xi + ((size_t)(n0 + lo16) * TOTAL_T + (size_t)t) * Mdim + kb;
      v4f f0 = *(const v4f*)(xr);
      v4f f1 = *(const v4f*)(xr + 4);
      v4f f2 = *(const v4f*)(xr + 16);          // elems 8..15 sit at K+16
      v4f f3 = *(const v4f*)(xr + 20);
      v16h a;
#pragma unroll
      for (int e = 0; e < 4; ++e) {
        a[e]      = (_Float16)f0[e];
        a[4 + e]  = (_Float16)f1[e];
        a[8 + e]  = (_Float16)f2[e];
        a[12 + e] = (_Float16)f3[e];
      }
      const _Float16* cr = Cb + (size_t)(h0 + lo16) * Mdim + mc * 32 + hi * 8;
      v16h b = pack16(*(const v8h*)cr, *(const v8h*)(cr + 16));
      acc = __builtin_amdgcn_wmma_f32_16x16x32_f16(
          false, a, false, b, (short)0, acc, false, false);
    }

    // ---- X_{t-1} A^T contribution: K = H = 512, sixteen K=32 chunks -----
#pragma unroll
    for (int kc = 0; kc < 16; ++kc) {
      const int kb = kc * 32 + (int)hi * 8;
      const _Float16* xrow = &Xbuf[p][lo16][0];
      v16h a = pack16(*(const v8h*)(xrow + kb),
                      *(const v8h*)(xrow + kb + 16));
      const _Float16* ar = Ab + (size_t)(h0 + lo16) * Hdim + kc * 32 + hi * 8;
      v16h b = pack16(*(const v8h*)ar, *(const v8h*)(ar + 16));
      acc = __builtin_amdgcn_wmma_f32_16x16x32_f16(
          false, a, false, b, (short)0, acc, false, false);
    }

    // ---- tanh + publish X_t (D-tile: col = lo16, row = j + hi*8) --------
#pragma unroll
    for (int j = 0; j < 8; ++j) {
      const int m = j + (int)hi * 8;            // batch row in tile
      Xbuf[p ^ 1][m][h0 + lo16] = (_Float16)tanhf(acc[j]);
    }
    __syncthreads();  // barrier 1: X_t visible, X_{t-1} reads retired

    // ---- Z_t = X_t W^T + eta_t, K split in half over waves 0..7 ---------
    v8f zacc = {};
    if (wave < 8) {
      const unsigned dt = wave & 3u;       // d-tile 0..3
      const unsigned d0 = dt * 16u;
      const int kc0 = (wave >> 2) ? 8 : 0; // K-half: chunks 0..7 / 8..15
#pragma unroll
      for (int kc = 0; kc < 8; ++kc) {
        const int kk = kc0 + kc;
        const int kb = kk * 32 + (int)hi * 8;
        const _Float16* xrow = &Xbuf[p ^ 1][lo16][0];
        v16h a = pack16(*(const v8h*)(xrow + kb),
                        *(const v8h*)(xrow + kb + 16));
        const _Float16* wr = Wb + (size_t)(d0 + lo16) * Hdim + kk * 32 + hi * 8;
        v16h b = pack16(*(const v8h*)wr, *(const v8h*)(wr + 16));
        zacc = __builtin_amdgcn_wmma_f32_16x16x32_f16(
            false, a, false, b, (short)0, zacc, false, false);
      }
      if (wave >= 4) {                      // publish upper-K partial
#pragma unroll
        for (int j = 0; j < 8; ++j)
          Zpart[dt][j + (int)hi * 8][lo16] = zacc[j];
      }
    }
    __syncthreads();  // barrier 2: partials visible

    if (wave < 4 && t >= WASHOUT_) {
#pragma unroll
      for (int j = 0; j < 8; ++j) {
        const int m = j + (int)hi * 8;
        const size_t n = n0 + (size_t)m;
        const float zsum = zacc[j] + Zpart[wave][m][lo16];
        const float e =
            eta[(n * TOTAL_T + (size_t)t) * Ddim + wave * 16u + lo16];
        out[(n * (size_t)T_OUT + (size_t)(t - WASHOUT_)) * Ddim +
            wave * 16u + lo16] = zsum + e;
      }
    }
  }
}

// ---------------------------------------------------------------------------
// Harness entry. Inputs: A(HxH), C(HxM), W(DxH), xi(N,T,M), eta(N,T,D),
// washout (scalar, compile-time constant here). Output: (N, T_OUT, D) f32.
// Workspace: 640KB of f16 weights.
// ---------------------------------------------------------------------------
extern "C" void kernel_launch(void* const* d_in, const int* in_sizes, int n_in,
                              void* d_out, int out_size, void* d_ws,
                              size_t ws_size, hipStream_t stream) {
  const float* A   = (const float*)d_in[0];
  const float* C   = (const float*)d_in[1];
  const float* W   = (const float*)d_in[2];
  const float* xi  = (const float*)d_in[3];
  const float* eta = (const float*)d_in[4];
  (void)in_sizes; (void)n_in; (void)out_size; (void)ws_size;

  _Float16* Af16 = (_Float16*)d_ws;
  _Float16* Cf16 = Af16 + (size_t)Hdim * Hdim;
  _Float16* Wf16 = Cf16 + (size_t)Hdim * Mdim;

  // 512*512 elements dominate: 1024 blocks x 256 threads covers all three.
  esn_convert_f16<<<(Hdim * Hdim + 255) / 256, 256, 0, stream>>>(
      A, C, W, Af16, Cf16, Wf16);

  esn_rollout<<<Nbatch / 16, 1024, 0, stream>>>(
      Af16, Cf16, Wf16, xi, eta, (float*)d_out);
}
